// SQNSTDPJelly_18743237279780
// MI455X (gfx1250) — compile-verified, matrix operand order checked by reading, not run on hardware
//
#include <hip/hip_runtime.h>
#include <hip/hip_bf16.h>

typedef __attribute__((ext_vector_type(16))) _Float16 v16h;
typedef __attribute__((ext_vector_type(8)))  _Float16 v8h;
typedef __attribute__((ext_vector_type(8)))  float    v8f;

#define BATCH 32
#define HIST  90
#define FEAT_N 50176            // 64*28*28
#define FC_IN (FEAT_N + HIST)   // 50266
#define FC_IN_PAD 50272         // multiple of 32

// ---------------- helpers ----------------
__device__ __forceinline__ unsigned f2ord(float f) {
  unsigned u = __float_as_uint(f);
  return (u & 0x80000000u) ? ~u : (u | 0x80000000u);
}
__device__ __forceinline__ float ord2f(unsigned u) {
  return __uint_as_float((u & 0x80000000u) ? (u & 0x7FFFFFFFu) : ~u);
}
__device__ __forceinline__ float rng01(unsigned idx, unsigned t) {
  unsigned x = idx * 0x9E3779B9u ^ (t * 0x85EBCA6Bu + 0xC2B2AE35u);
  x ^= x >> 16; x *= 0x7FEB352Du;
  x ^= x >> 15; x *= 0x846CA68Bu;
  x ^= x >> 16;
  return (float)(x >> 8) * (1.0f / 16777216.0f);
}
__device__ __forceinline__ v16h cat8(v8h lo, v8h hi) {
  return __builtin_shufflevector(lo, hi, 0,1,2,3,4,5,6,7,8,9,10,11,12,13,14,15);
}

// ---------------- DoG depthwise conv + global min/max ----------------
__global__ __launch_bounds__(256)
void dog_minmax_kernel(const float* __restrict__ state, const float* __restrict__ dogw,
                       float* __restrict__ xdog, unsigned* __restrict__ mm) {
  const long tot = (long)BATCH * 3 * 224 * 224;
  long i = (long)blockIdx.x * blockDim.x + threadIdx.x;
  bool act = (i < tot);
  float val = 0.f;
  if (act) {
    int x = (int)(i % 224); long r = i / 224;
    int y = (int)(r % 224); r /= 224;
    int c = (int)(r % 3);   int b = (int)(r / 3);
    const float* sp = state + ((long)b * 3 + c) * 224 * 224;
    const float* wp = dogw + c * 49;
    #pragma unroll
    for (int kh = 0; kh < 7; ++kh) {
      int iy = y + kh - 3;
      if (iy < 0 || iy >= 224) continue;
      #pragma unroll
      for (int kw = 0; kw < 7; ++kw) {
        int ix = x + kw - 3;
        if (ix < 0 || ix >= 224) continue;
        val += sp[iy * 224 + ix] * wp[kh * 7 + kw];
      }
    }
    xdog[i] = val;
  }
  __shared__ float smin[256], smax[256];
  smin[threadIdx.x] = act ? val :  3.4e38f;
  smax[threadIdx.x] = act ? val : -3.4e38f;
  __syncthreads();
  for (int s = 128; s > 0; s >>= 1) {
    if ((int)threadIdx.x < s) {
      smin[threadIdx.x] = fminf(smin[threadIdx.x], smin[threadIdx.x + s]);
      smax[threadIdx.x] = fmaxf(smax[threadIdx.x], smax[threadIdx.x + s]);
    }
    __syncthreads();
  }
  if (threadIdx.x == 0) {
    atomicMin(&mm[0], f2ord(smin[0]));
    atomicMax(&mm[1], f2ord(smax[0]));
  }
}

// ---------------- Poisson encoder (normalization fused) ----------------
__global__ __launch_bounds__(256)
void encode_kernel(const float* __restrict__ xdog, const unsigned* __restrict__ mm,
                   _Float16* __restrict__ spk, unsigned t) {
  const long tot = (long)BATCH * 3 * 224 * 224;
  long i = (long)blockIdx.x * blockDim.x + threadIdx.x;
  if (i >= tot) return;
  float mn = ord2f(mm[0]), mx = ord2f(mm[1]);
  float xn = (xdog[i] - mn) / (mx - mn + 1e-5f);
  float u = rng01((unsigned)i, t);
  spk[i] = (_Float16)((u <= xn) ? 1.0f : 0.0f);
}

// ---------------- implicit-GEMM conv + fused IF neuron (WMMA f16) ------
// Block: one (batch, oc-tile of 16, strip of PIXB pixels). im2col strip and the
// 16 weight rows are staged in LDS once; fragments are pure ds_load_b128 pairs.
template<int CIN, int KS, int PAD, int H, int W, int OC, int PIXB>
__global__ __launch_bounds__((PIXB / 16) * 32)
void conv_if_kernel(const _Float16* __restrict__ spk_in,
                    const float*   __restrict__ wgt,
                    float*         __restrict__ vmem,
                    _Float16*      __restrict__ spk_out) {
  constexpr int KREAL = CIN * KS * KS;
  constexpr int KPAD  = (KREAL + 31) & ~31;
  constexpr int NPIX  = H * W;
  constexpr int OCT   = OC / 16;
  constexpr int NW    = PIXB / 16;           // waves per block
  constexpr int NT    = NW * 32;             // threads per block
  constexpr int LOGP  = (PIXB == 128) ? 7 : 6;

  __shared__ __align__(16) _Float16 wlds[16 * KPAD];    // this block's oc rows
  __shared__ __align__(16) _Float16 clds[PIXB * KPAD];  // im2col strip

  const int tid = threadIdx.x;
  int blk = blockIdx.x;
  int pb = blk % (NPIX / PIXB); int r = blk / (NPIX / PIXB);
  int ot = r % OCT;             int b = r / OCT;
  const int pix0 = pb * PIXB;

  // stage 16 weight rows (fp32 -> fp16, zero-pad K)
  for (int row = 0; row < 16; ++row) {
    const float* wrow = wgt + (long)(ot * 16 + row) * KREAL;
    for (int k = tid; k < KPAD; k += NT)
      wlds[row * KPAD + k] = (k < KREAL) ? (_Float16)wrow[k] : (_Float16)0.0f;
  }

  // im2col fill: idx = k*PIXB + p so k is wave-uniform, p coalesced
  const _Float16* in_b = spk_in + (long)b * CIN * NPIX;
  for (int idx = tid; idx < KPAD * PIXB; idx += NT) {
    int p = idx & (PIXB - 1);
    int k = idx >> LOGP;
    _Float16 v = (_Float16)0.0f;
    if (k < KREAL) {
      int ic = k / (KS * KS); int rem = k - ic * (KS * KS);
      int kh = rem / KS;      int kw = rem - kh * KS;
      int n = pix0 + p;
      int y = n / W, x = n - y * W;
      int iy = y + kh - PAD, ix = x + kw - PAD;
      if ((unsigned)iy < (unsigned)H && (unsigned)ix < (unsigned)W)
        v = in_b[(long)ic * NPIX + iy * W + ix];
    }
    clds[p * KPAD + k] = v;
  }
  __syncthreads();

  const int wave = tid >> 5;
  const int lane = tid & 31;
  const int mr = lane & 15;            // A row / D column-half selector
  const int kb = (lane >> 4) * 8;      // K sub-offset per half-wave

  const _Float16* arow = &wlds[mr * KPAD + kb];
  const _Float16* brow = &clds[(wave * 16 + (lane & 15)) * KPAD + kb];

  v8f acc = {};
  #pragma unroll
  for (int kc = 0; kc < KPAD / 32; ++kc) {
    const int k0 = kc * 32;
    v16h a  = cat8(*(const v8h*)(arow + k0), *(const v8h*)(arow + k0 + 16));
    v16h bf = cat8(*(const v8h*)(brow + k0), *(const v8h*)(brow + k0 + 16));
    acc = __builtin_amdgcn_wmma_f32_16x16x32_f16(false, a, false, bf,
                                                 (short)0, acc, false, false);
  }

  // Fused IF neuron: integrate, threshold at 1.0, hard reset to 0.
  const int n = pix0 + wave * 16 + (lane & 15);
  #pragma unroll
  for (int r8 = 0; r8 < 8; ++r8) {
    int oc = ot * 16 + r8 + (lane >> 4) * 8;
    long idx = ((long)b * OC + oc) * NPIX + n;
    float v = vmem[idx] + acc[r8];
    float s = (v >= 1.0f) ? 1.0f : 0.0f;
    vmem[idx] = v * (1.0f - s);
    spk_out[idx] = (_Float16)s;
  }
}

// ---------------- 2x2 max pool on fp16 spike maps ----------------
template<int C, int H, int W>
__global__ __launch_bounds__(256)
void pool_kernel(const _Float16* __restrict__ in, _Float16* __restrict__ out) {
  constexpr int OH = H / 2, OW = W / 2;
  long i = (long)blockIdx.x * blockDim.x + threadIdx.x;
  const long tot = (long)BATCH * C * OH * OW;
  if (i >= tot) return;
  int ox = (int)(i % OW); long r = i / OW;
  int oy = (int)(r % OH); r /= OH;            // r = b*C + c
  const _Float16* p = in + r * (long)H * W + (2 * oy) * W + 2 * ox;
  float m = fmaxf(fmaxf((float)p[0], (float)p[1]),
                  fmaxf((float)p[W], (float)p[W + 1]));
  out[i] = (_Float16)m;
}

// ---------------- feature assembly: [spk3p | history] -> fp16, padded --
__global__ __launch_bounds__(256)
void feat_kernel(const _Float16* __restrict__ spk3p, const float* __restrict__ history,
                 _Float16* __restrict__ feat) {
  const long tot = (long)BATCH * FC_IN_PAD;
  long i = (long)blockIdx.x * blockDim.x + threadIdx.x;
  if (i >= tot) return;
  int k = (int)(i % FC_IN_PAD); int b = (int)(i / FC_IN_PAD);
  _Float16 v = (_Float16)0.0f;
  if (k < FEAT_N)      v = spk3p[(long)b * FEAT_N + k];
  else if (k < FC_IN)  v = (_Float16)history[b * HIST + (k - FEAT_N)];
  feat[i] = v;
}

// ---------------- FC layer via WMMA: out(32,N) = relu(X(32,KPAD) W(N,K)^T + b)
template<int N, int K, int KPAD, bool RELU>
__global__ __launch_bounds__(256)
void fc_wmma_kernel(const _Float16* __restrict__ X, const float* __restrict__ Wt,
                    const float* __restrict__ bias, _Float16* __restrict__ out) {
  int wave = threadIdx.x >> 5;
  int lane = threadIdx.x & 31;
  int gw = blockIdx.x * (blockDim.x >> 5) + wave;   // waves = 2 * (N/16)
  int mt = gw % 2, nt = gw / 2;
  int mr = lane & 15;
  int nn = nt * 16 + (lane & 15);
  int kb = (lane >> 4) * 8;

  const _Float16* xrow = X + (long)(mt * 16 + mr) * KPAD + kb;
  const float*    wrow = Wt + (long)nn * K;

  v8f acc = {};
  #pragma unroll 1
  for (int k0 = 0; k0 < KPAD; k0 += 32) {
    v16h a = cat8(*(const v8h*)(xrow + k0), *(const v8h*)(xrow + k0 + 16));
    v16h bf;
    #pragma unroll
    for (int i = 0; i < 8; ++i) {
      int klo = k0 + kb + i;
      int khi = k0 + kb + 16 + i;
      bf[i]     = (_Float16)((klo < K) ? wrow[klo] : 0.0f);
      bf[i + 8] = (_Float16)((khi < K) ? wrow[khi] : 0.0f);
    }
    acc = __builtin_amdgcn_wmma_f32_16x16x32_f16(false, a, false, bf,
                                                 (short)0, acc, false, false);
  }
  #pragma unroll
  for (int r = 0; r < 8; ++r) {
    int m = mt * 16 + r + (lane >> 4) * 8;
    int c = nt * 16 + (lane & 15);
    float v = acc[r] + bias[c];
    if (RELU) v = fmaxf(v, 0.0f);
    out[(long)m * N + c] = (_Float16)v;
  }
}

// ---------------- FC3 (9 outputs): tiny VALU dot products ----------------
__global__ __launch_bounds__(288)
void fc3_kernel(const _Float16* __restrict__ h2, const float* __restrict__ W3,
                const float* __restrict__ b3, float* __restrict__ q) {
  int i = blockIdx.x * blockDim.x + threadIdx.x;
  if (i >= BATCH * 9) return;
  int o = i % 9, b = i / 9;
  float s = b3[o];
  #pragma unroll 4
  for (int k = 0; k < 256; ++k)
    s += (float)h2[b * 256 + k] * W3[o * 256 + k];
  q[i] = s;
}

// ---------------- launch ----------------
static size_t aloc(size_t& cur, size_t bytes) {
  size_t off = (cur + 255) & ~(size_t)255;
  cur = off + bytes;
  return off;
}

extern "C" void kernel_launch(void* const* d_in, const int* in_sizes, int n_in,
                              void* d_out, int out_size, void* d_ws, size_t ws_size,
                              hipStream_t stream) {
  (void)in_sizes; (void)n_in; (void)out_size; (void)ws_size;
  const float* state   = (const float*)d_in[0];
  const float* history = (const float*)d_in[1];
  const float* dog_w   = (const float*)d_in[2];
  const float* conv1_w = (const float*)d_in[3];
  const float* conv2_w = (const float*)d_in[4];
  const float* conv3_w = (const float*)d_in[5];
  const float* fc1_w   = (const float*)d_in[6];
  const float* fc1_b   = (const float*)d_in[7];
  const float* fc2_w   = (const float*)d_in[8];
  const float* fc2_b   = (const float*)d_in[9];
  const float* fc3_w   = (const float*)d_in[10];
  const float* fc3_b   = (const float*)d_in[11];
  float* q = (float*)d_out;
  char* ws = (char*)d_ws;

  // workspace layout
  size_t cur = 0;
  size_t o_xdog  = aloc(cur, (size_t)BATCH * 3 * 224 * 224 * 4);
  size_t o_mm    = aloc(cur, 8);
  size_t o_v1    = aloc(cur, (size_t)BATCH * 16 * 224 * 224 * 4);
  size_t o_v2    = aloc(cur, (size_t)BATCH * 32 * 112 * 112 * 4);
  size_t o_v3    = aloc(cur, (size_t)BATCH * 64 * 56  * 56  * 4);
  size_t o_spkin = aloc(cur, (size_t)BATCH * 3  * 224 * 224 * 2);
  size_t o_spk1  = aloc(cur, (size_t)BATCH * 16 * 224 * 224 * 2);
  size_t o_spk1p = aloc(cur, (size_t)BATCH * 16 * 112 * 112 * 2);
  size_t o_spk2  = aloc(cur, (size_t)BATCH * 32 * 112 * 112 * 2);
  size_t o_spk2p = aloc(cur, (size_t)BATCH * 32 * 56  * 56  * 2);
  size_t o_spk3  = aloc(cur, (size_t)BATCH * 64 * 56  * 56  * 2);
  size_t o_spk3p = aloc(cur, (size_t)BATCH * 64 * 28  * 28  * 2);
  size_t o_feat  = aloc(cur, (size_t)BATCH * FC_IN_PAD * 2);
  size_t o_h1    = aloc(cur, (size_t)BATCH * 128 * 2);
  size_t o_h2    = aloc(cur, (size_t)BATCH * 256 * 2);

  float*     xdog  = (float*)(ws + o_xdog);
  unsigned*  mm    = (unsigned*)(ws + o_mm);
  float*     v1    = (float*)(ws + o_v1);
  float*     v2    = (float*)(ws + o_v2);
  float*     v3    = (float*)(ws + o_v3);
  _Float16*  spkin = (_Float16*)(ws + o_spkin);
  _Float16*  spk1  = (_Float16*)(ws + o_spk1);
  _Float16*  spk1p = (_Float16*)(ws + o_spk1p);
  _Float16*  spk2  = (_Float16*)(ws + o_spk2);
  _Float16*  spk2p = (_Float16*)(ws + o_spk2p);
  _Float16*  spk3  = (_Float16*)(ws + o_spk3);
  _Float16*  spk3p = (_Float16*)(ws + o_spk3p);
  _Float16*  feat  = (_Float16*)(ws + o_feat);
  _Float16*  h1    = (_Float16*)(ws + o_h1);
  _Float16*  h2    = (_Float16*)(ws + o_h2);

  // init: min = +inf(ordered 0xFF..), max = 0x00.. ; membranes = 0 (v1..v3 contiguous)
  hipMemsetAsync(mm, 0xFF, 4, stream);
  hipMemsetAsync(mm + 1, 0x00, 4, stream);
  hipMemsetAsync(v1, 0, (o_v3 + (size_t)BATCH * 64 * 56 * 56 * 4) - o_v1, stream);

  // DoG + min/max
  {
    long tot = (long)BATCH * 3 * 224 * 224;
    dog_minmax_kernel<<<(int)((tot + 255) / 256), 256, 0, stream>>>(state, dog_w, xdog, mm);
  }

  const int T = 20;
  for (int t = 0; t < T; ++t) {
    { // encode
      long tot = (long)BATCH * 3 * 224 * 224;
      encode_kernel<<<(int)((tot + 255) / 256), 256, 0, stream>>>(xdog, mm, spkin, (unsigned)t);
    }
    { // conv1 + IF : blocks = 32*1*(50176/128) = 12544, 8 waves each
      conv_if_kernel<3, 5, 2, 224, 224, 16, 128>
        <<<12544, 256, 0, stream>>>(spkin, conv1_w, v1, spk1);
      long tot = (long)BATCH * 16 * 112 * 112;
      pool_kernel<16, 224, 224><<<(int)((tot + 255) / 256), 256, 0, stream>>>(spk1, spk1p);
    }
    { // conv2 + IF : blocks = 32*2*(12544/128) = 6272, 8 waves each
      conv_if_kernel<16, 3, 1, 112, 112, 32, 128>
        <<<6272, 256, 0, stream>>>(spk1p, conv2_w, v2, spk2);
      long tot = (long)BATCH * 32 * 56 * 56;
      pool_kernel<32, 112, 112><<<(int)((tot + 255) / 256), 256, 0, stream>>>(spk2, spk2p);
    }
    { // conv3 + IF : blocks = 32*4*(3136/64) = 6272, 4 waves each
      conv_if_kernel<32, 3, 1, 56, 56, 64, 64>
        <<<6272, 128, 0, stream>>>(spk2p, conv3_w, v3, spk3);
      if (t == T - 1) {  // only the last step's pooled spikes feed the FC head
        long tot = (long)BATCH * 64 * 28 * 28;
        pool_kernel<64, 56, 56><<<(int)((tot + 255) / 256), 256, 0, stream>>>(spk3, spk3p);
      }
    }
  }

  { // features
    long tot = (long)BATCH * FC_IN_PAD;
    feat_kernel<<<(int)((tot + 255) / 256), 256, 0, stream>>>(spk3p, history, feat);
  }
  // FC1: waves = 2*(128/16) = 16 -> 2 blocks
  fc_wmma_kernel<128, FC_IN, FC_IN_PAD, true><<<2, 256, 0, stream>>>(feat, fc1_w, fc1_b, h1);
  // FC2: waves = 2*(256/16) = 32 -> 4 blocks
  fc_wmma_kernel<256, 128, 128, true><<<4, 256, 0, stream>>>(h1, fc2_w, fc2_b, h2);
  // FC3
  fc3_kernel<<<1, 288, 0, stream>>>(h2, fc3_w, fc3_b, q);
}
